// Net_16561393893885
// MI455X (gfx1250) — matrix operand order repaired and checked by reading, not compiled, not run on hardware
//
#include <hip/hip_runtime.h>
#include <hip/hip_bf16.h>
#include <math.h>

#define GG 3
#define NN 40000
#define FF 256
#define HH 128
#define EE 640000

typedef __attribute__((ext_vector_type(16))) __bf16 v16bf;
typedef __attribute__((ext_vector_type(8)))  __bf16 v8bf;
typedef __attribute__((ext_vector_type(4)))  __bf16 v4bf;
typedef __attribute__((ext_vector_type(8)))  float  v8f;

__device__ __forceinline__ __bf16 f2bf(float f) {
  unsigned u = __builtin_bit_cast(unsigned, f);
  u += 0x7FFFu + ((u >> 16) & 1u);            // round-to-nearest-even
  unsigned short h = (unsigned short)(u >> 16);
  return __builtin_bit_cast(__bf16, h);
}

// ---- f32 -> bf16 conversion, 4 elements/thread (b128 read, b64 write) ----
__global__ __launch_bounds__(256) void cvt_f32_bf16_v4(const float* __restrict__ in,
                                                       __bf16* __restrict__ out,
                                                       size_t n4) {
  size_t i = (size_t)blockIdx.x * blockDim.x + threadIdx.x;
  if (i >= n4) return;
  float4 v = ((const float4*)in)[i];
  v4bf o;
  o[0] = f2bf(v.x); o[1] = f2bf(v.y); o[2] = f2bf(v.z); o[3] = f2bf(v.w);
  ((v4bf*)out)[i] = o;
}

// ---- weight convert + transpose: Wt[n*K + k] = bf16(W[k*Nc + n]) ----
__global__ __launch_bounds__(256) void cvt_transpose_w(const float* __restrict__ W,
                                                       __bf16* __restrict__ Wt,
                                                       int K, int Nc) {
  int idx = blockIdx.x * blockDim.x + threadIdx.x;
  if (idx >= K * Nc) return;
  int k = idx / Nc, n = idx % Nc;
  Wt[(size_t)n * K + k] = f2bf(W[idx]);
}

// out(MxNc,f32) = A(MxK,bf16) * Wt(NcxK,bf16 transposed) [+ bias]
// One wave computes a 16(M) x 64(N) strip: 1 A-fragment feeds 4 WMMAs per K-step.
// M%16==0, K%32==0, Nc%64==0 for all call sites here.
__global__ __launch_bounds__(32) void gemm_wmma_bf16(
    const __bf16* __restrict__ A, const __bf16* __restrict__ Wt,
    const float* __restrict__ bias, float* __restrict__ out,
    int M, int K, int Nc) {
  const int tm   = blockIdx.x;
  const int tn   = blockIdx.y;           // 64-wide N strip
  const int lane = threadIdx.x;
  const int half = lane >> 4;
  const int ml   = lane & 15;
  const int row  = tm * 16 + ml;

  v8f acc[4];
#pragma unroll
  for (int s = 0; s < 4; ++s) acc[s] = (v8f){0.f,0.f,0.f,0.f,0.f,0.f,0.f,0.f};

  // A fragment element i -> K = k0 + (i>>3)*16 + half*8 + (i&7)
  const __bf16* arow  = A + (size_t)row * K + half * 8;
  // B fragment element i -> K = k0 + (i>>3)*16 + half*8 + (i&7), for column ncol (row of Wt)
  const __bf16* wbase = Wt + ((size_t)tn * 64 + ml) * K + half * 8;

  for (int k0 = 0; k0 < K; k0 += 32) {
    v8bf a0 = *(const v8bf*)(arow + k0);
    v8bf a1 = *(const v8bf*)(arow + k0 + 16);
    v16bf a = __builtin_shufflevector(a0, a1, 0,1,2,3,4,5,6,7,8,9,10,11,12,13,14,15);
#pragma unroll
    for (int s = 0; s < 4; ++s) {
      const __bf16* wp = wbase + (size_t)s * 16 * K + k0;
      v8bf b0 = *(const v8bf*)wp;
      v8bf b1 = *(const v8bf*)(wp + 16);
      v16bf b = __builtin_shufflevector(b0, b1, 0,1,2,3,4,5,6,7,8,9,10,11,12,13,14,15);
      acc[s] = __builtin_amdgcn_wmma_f32_16x16x32_bf16(
          false, a, false, b, (short)0, acc[s], false, false);
    }
  }

#pragma unroll
  for (int s = 0; s < 4; ++s) {
    const int ncol = tn * 64 + s * 16 + ml;
    const float bv = bias ? bias[ncol] : 0.0f;
#pragma unroll
    for (int r = 0; r < 8; ++r) {
      const int m = tm * 16 + r + 8 * half;  // C/D layout: VGPR r = row r (+8 upper half)
      out[(size_t)m * Nc + ncol] = acc[s][r] + bv;
    }
  }
}

__global__ __launch_bounds__(256) void fill_f32(float* __restrict__ p, size_t n, float v) {
  size_t i = (size_t)blockIdx.x * blockDim.x + threadIdx.x;
  if (i < n) p[i] = v;
}

__global__ __launch_bounds__(256) void deg_accum(const int* __restrict__ cols,
                                                 float* __restrict__ deg, int E) {
  int e = blockIdx.x * blockDim.x + threadIdx.x;
  if (e < E)
    __hip_atomic_fetch_add(&deg[cols[e]], 1.0f, __ATOMIC_RELAXED,
                           __HIP_MEMORY_SCOPE_AGENT);
}

__global__ __launch_bounds__(256) void make_dinv(float* __restrict__ d, int n) {
  int i = blockIdx.x * blockDim.x + threadIdx.x;
  if (i < n) { float v = d[i]; d[i] = (v > 0.f) ? rsqrtf(v) : 0.f; }
}

// acc[col] += dinv[row]*dinv[col] * xw[row, :].  32 lanes x float4 cover H=128.
__global__ __launch_bounds__(256) void scatter_edges(
    const int* __restrict__ rows, const int* __restrict__ cols,
    const float* __restrict__ dinv, const float* __restrict__ xw,
    float* __restrict__ acc, int E) {
  size_t gid = (size_t)blockIdx.x * blockDim.x + threadIdx.x;
  int e    = (int)(gid >> 5);
  int lane = (int)(gid & 31);
  if (e >= E) return;
  int r = rows[e], c = cols[e];
  float norm = dinv[r] * dinv[c];
  int f0 = lane * 4;
  float4 v = *(const float4*)(xw + (size_t)r * HH + f0);
  float* dst = acc + (size_t)c * HH + f0;
  __hip_atomic_fetch_add(dst + 0, norm * v.x, __ATOMIC_RELAXED, __HIP_MEMORY_SCOPE_AGENT);
  __hip_atomic_fetch_add(dst + 1, norm * v.y, __ATOMIC_RELAXED, __HIP_MEMORY_SCOPE_AGENT);
  __hip_atomic_fetch_add(dst + 2, norm * v.z, __ATOMIC_RELAXED, __HIP_MEMORY_SCOPE_AGENT);
  __hip_atomic_fetch_add(dst + 3, norm * v.w, __ATOMIC_RELAXED, __HIP_MEMORY_SCOPE_AGENT);
}

// self-loop term + bias: acc[i,f] += dinv[i]^2 * xw[i,f] + bias[f]
__global__ __launch_bounds__(256) void gcn_finish(
    const float* __restrict__ xw, const float* __restrict__ dinv,
    const float* __restrict__ bias, float* __restrict__ acc, int n) {
  size_t idx = (size_t)blockIdx.x * blockDim.x + threadIdx.x;
  if (idx >= (size_t)n * HH) return;
  int i = (int)(idx / HH);
  int f = (int)(idx % HH);
  float di = dinv[i];
  acc[idx] += di * di * xw[idx] + bias[f];
}

__global__ __launch_bounds__(256) void neg_add(const float* __restrict__ a,
                                               const float* __restrict__ b,
                                               float* __restrict__ o, size_t n) {
  size_t i = (size_t)blockIdx.x * blockDim.x + threadIdx.x;
  if (i < n) o[i] = -(a[i] + b[i]);
}

__global__ __launch_bounds__(256) void add2(const float* __restrict__ a,
                                            const float* __restrict__ b,
                                            float* __restrict__ o, size_t n) {
  size_t i = (size_t)blockIdx.x * blockDim.x + threadIdx.x;
  if (i < n) o[i] = a[i] + b[i];
}

__global__ __launch_bounds__(256) void relu_copy(const float* __restrict__ a,
                                                 float* __restrict__ o, size_t n) {
  size_t i = (size_t)blockIdx.x * blockDim.x + threadIdx.x;
  if (i < n) o[i] = fmaxf(a[i], 0.0f);
}

// in-place row softmax, one wave32 per row (F=256 -> 8 elems/lane)
__global__ __launch_bounds__(256) void softmax_rows(float* __restrict__ x, int n, int F) {
  int row  = blockIdx.x * (blockDim.x / 32) + (threadIdx.x / 32);
  int lane = threadIdx.x & 31;
  if (row >= n) return;
  float* p = x + (size_t)row * F;
  float mx = -3.402823466e38f;
  for (int j = lane; j < F; j += 32) mx = fmaxf(mx, p[j]);
  for (int o = 16; o > 0; o >>= 1) mx = fmaxf(mx, __shfl_xor(mx, o, 32));
  float s = 0.f;
  for (int j = lane; j < F; j += 32) { float e = __expf(p[j] - mx); p[j] = e; s += e; }
  for (int o = 16; o > 0; o >>= 1) s += __shfl_xor(s, o, 32);
  float inv = 1.0f / s;
  for (int j = lane; j < F; j += 32) p[j] *= inv;
}

extern "C" void kernel_launch(void* const* d_in, const int* in_sizes, int n_in,
                              void* d_out, int out_size, void* d_ws, size_t ws_size,
                              hipStream_t stream) {
  (void)in_sizes; (void)n_in; (void)out_size; (void)ws_size;
  const float* x    = (const float*)d_in[0];
  const int*   edge = (const int*)d_in[1];      // (G,2,E)
  const float* fc1W = (const float*)d_in[2];
  const float* fc1b = (const float*)d_in[3];
  const float* c1W  = (const float*)d_in[4];
  const float* c1b  = (const float*)d_in[5];
  const float* c2W  = (const float*)d_in[6];
  const float* c2b  = (const float*)d_in[7];
  const float* d1W  = (const float*)d_in[8];
  const float* d1b  = (const float*)d_in[9];
  const float* d2W  = (const float*)d_in[10];
  const float* d2b  = (const float*)d_in[11];
  const float* fc2W = (const float*)d_in[12];
  const float* fc2b = (const float*)d_in[13];

  const size_t NH = (size_t)NN * HH;
  float* out = (float*)d_out;
  float* pre = out;                       // (G,N,H)
  float* enc = out + (size_t)GG * NH;     // (G,N,H)
  float* hst = enc + (size_t)GG * NH;     // (G,N,H)
  float* fin = hst + (size_t)GG * NH;     // (N,H)
  float* los = fin + NH;                  // (N,F)

  float*  wsf  = (float*)d_ws;
  float*  dinv = wsf;                      // G*N f32
  float*  bufA = wsf + (size_t)GG * NN;    // N*H f32 each
  float*  bufB = bufA + NH;
  float*  bufC = bufB + NH;
  __bf16* Abf  = (__bf16*)(bufC + NH);     // N*F bf16 (max activation)
  __bf16* Wbf  = Abf + (size_t)NN * FF;    // 32K bf16 (max weight)

  dim3 blk256(256);
  auto cdiv = [](size_t a, size_t b) { return (unsigned)((a + b - 1) / b); };

  // GEMM front-end: convert A once, convert+transpose W once, then WMMA GEMM.
  auto gemmF = [&](const float* Af, const float* Wf, const float* b, float* o,
                   int M, int K, int Nc) {
    size_t n4 = (size_t)M * K / 4;
    cvt_f32_bf16_v4<<<cdiv(n4, 256), blk256, 0, stream>>>(Af, Abf, n4);
    cvt_transpose_w<<<cdiv((size_t)K * Nc, 256), blk256, 0, stream>>>(Wf, Wbf, K, Nc);
    gemm_wmma_bf16<<<dim3(M / 16, Nc / 64), 32, 0, stream>>>(Abf, Wbf, b, o, M, K, Nc);
  };

  // ---- degrees + dinv for all 3 graphs (self-loop => deg starts at 1) ----
  for (int g = 0; g < GG; ++g) {
    const int* cols = edge + (size_t)g * 2 * EE + EE;
    float* dg = dinv + (size_t)g * NN;
    fill_f32<<<cdiv(NN, 256), blk256, 0, stream>>>(dg, (size_t)NN, 1.0f);
    deg_accum<<<cdiv(EE, 256), blk256, 0, stream>>>(cols, dg, EE);
    make_dinv<<<cdiv(NN, 256), blk256, 0, stream>>>(dg, NN);
  }

  // one GCN conv: accb = Anorm * (in @ W) + bias   (accb must differ from in/xw)
  auto gcn = [&](const float* in, const float* W, const float* b, int g,
                 float* xw, float* accb) {
    gemmF(in, W, nullptr, xw, NN, HH, HH);
    fill_f32<<<cdiv(NH, 256), blk256, 0, stream>>>(accb, NH, 0.0f);
    const int* rows = edge + (size_t)g * 2 * EE;
    const int* cols = rows + EE;
    scatter_edges<<<cdiv((size_t)EE * 32, 256), blk256, 0, stream>>>(
        rows, cols, dinv + (size_t)g * NN, xw, accb, EE);
    gcn_finish<<<cdiv(NH, 256), blk256, 0, stream>>>(xw, dinv + (size_t)g * NN, b,
                                                     accb, NN);
  };

  // ---- encoders ----
  for (int g = 0; g < GG; ++g) {
    gemmF(x + (size_t)g * NN * FF, fc1W + (size_t)g * FF * HH,
          fc1b + (size_t)g * HH, pre + (size_t)g * NH, NN, FF, HH);
    gcn(pre + (size_t)g * NH, c1W + (size_t)g * HH * HH, c1b + (size_t)g * HH, g,
        bufB, bufC);
    gcn(bufC, c2W + (size_t)g * HH * HH, c2b + (size_t)g * HH, g, bufB, bufA);
    relu_copy<<<cdiv(NH, 256), blk256, 0, stream>>>(bufA, enc + (size_t)g * NH, NH);
  }

  // ---- cross-graph sequential block ----
  // i=0: h0 = dgcn0( -(enc0+enc1) )
  neg_add<<<cdiv(NH, 256), blk256, 0, stream>>>(enc, enc + NH, bufA, NH);
  gcn(bufA, d1W, d1b, 0, bufB, bufC);
  gcn(bufC, d2W, d2b, 0, bufB, hst);
  // i=1: h1 = dgcn1( -(enc1+h0) )
  neg_add<<<cdiv(NH, 256), blk256, 0, stream>>>(enc + NH, hst, bufA, NH);
  gcn(bufA, d1W + (size_t)HH * HH, d1b + HH, 1, bufB, bufC);
  gcn(bufC, d2W + (size_t)HH * HH, d2b + HH, 1, bufB, hst + NH);
  // i=2: h2 = dgcn2( 0 )
  fill_f32<<<cdiv(NH, 256), blk256, 0, stream>>>(bufA, NH, 0.0f);
  gcn(bufA, d1W + (size_t)2 * HH * HH, d1b + 2 * HH, 2, bufB, bufC);
  gcn(bufC, d2W + (size_t)2 * HH * HH, d2b + 2 * HH, 2, bufB, hst + 2 * NH);

  // ---- fin = h0 + h1 ; loss = softmax(fin @ fc2_W + fc2_b) ----
  add2<<<cdiv(NH, 256), blk256, 0, stream>>>(hst, hst + NH, fin, NH);
  gemmF(fin, fc2W, fc2b, los, NN, HH, FF);
  softmax_rows<<<cdiv((size_t)NN * 32, 256), blk256, 0, stream>>>(los, NN, FF);
}
